// Block_32796370273139
// MI455X (gfx1250) — matrix operand order repaired and checked by reading, not compiled, not run on hardware
//
#include <hip/hip_runtime.h>
#include <hip/hip_bf16.h>
#include <math.h>

typedef __attribute__((ext_vector_type(16))) _Float16 v16h;
typedef __attribute__((ext_vector_type(8)))  _Float16 v8h;
typedef __attribute__((ext_vector_type(8)))  float    v8f;

#define BB   2
#define LL   1024
#define EE   1024
#define DIN  2048
#define NN   16
#define RR   64
#define KK   4
#define ROWS (BB*LL)   /* 2048 */
#define XDBL_LD 128    /* x_dbl row stride (96 padded to 128) */

// ---------------------------------------------------------------- utilities
__global__ void cvt_f32_f16(const float* __restrict__ s, _Float16* __restrict__ d, int n) {
    int i = blockIdx.x * blockDim.x + threadIdx.x;
    if (i < n) d[i] = (_Float16)s[i];
}

// x_proj weight (96,2048) -> f16 (128,2048), rows 96..127 zero.
__global__ void cvt_pad_w2(const float* __restrict__ s, _Float16* __restrict__ d) {
    int idx = blockIdx.x * blockDim.x + threadIdx.x;
    if (idx >= XDBL_LD * DIN) return;
    int row = idx >> 11;   // /2048
    d[idx] = (row < 96) ? (_Float16)s[idx] : (_Float16)0.f;
}

// ---------------------------------------------------------------- layernorm
__global__ void ln_kernel(const float* __restrict__ X, const float* __restrict__ Rs,
                          const float* __restrict__ w, const float* __restrict__ bb,
                          _Float16* __restrict__ out16, float* __restrict__ sum_out) {
    __shared__ float red[256];
    int row = blockIdx.x;
    const float* xr = X + (size_t)row * EE;
    float v[4];
    float s = 0.f;
#pragma unroll
    for (int j = 0; j < 4; j++) {
        int i = threadIdx.x + 256 * j;
        float t = xr[i];
        if (Rs) t += Rs[(size_t)row * EE + i];
        if (sum_out) sum_out[(size_t)row * EE + i] = t;
        v[j] = t; s += t;
    }
    red[threadIdx.x] = s; __syncthreads();
    for (int o = 128; o > 0; o >>= 1) {
        if (threadIdx.x < o) red[threadIdx.x] += red[threadIdx.x + o];
        __syncthreads();
    }
    float mean = red[0] * (1.f / EE);
    __syncthreads();
    float s2 = 0.f;
#pragma unroll
    for (int j = 0; j < 4; j++) { float c = v[j] - mean; s2 += c * c; }
    red[threadIdx.x] = s2; __syncthreads();
    for (int o = 128; o > 0; o >>= 1) {
        if (threadIdx.x < o) red[threadIdx.x] += red[threadIdx.x + o];
        __syncthreads();
    }
    float rstd = rsqrtf(red[0] * (1.f / EE) + 1e-5f);
#pragma unroll
    for (int j = 0; j < 4; j++) {
        int i = threadIdx.x + 256 * j;
        out16[(size_t)row * EE + i] = (_Float16)((v[j] - mean) * rstd * w[i] + bb[i]);
    }
}

// ---------------------------------------------------------------- WMMA GEMM
// C[M,N] (f32) = A[M,K](f16 row-major) * Bw[N,K](f16 row-major)   (C = A*B^T)
// Requirements: M%32==0, N%64==0, K%32==0 (all call sites satisfy these).
// One wave -> 32x64 strip (2 M-tiles x 4 N-tiles, 8 WMMAs / k-step).
// Single fragment set (no spills); latency hidden with global_prefetch_b8 of
// the cache lines two k-steps (128B) ahead — speculative, zero VGPR cost,
// OOB prefetches are dropped by hardware. EXEC stays all-ones for WMMA.
__global__ void wmma_gemm_nt(const _Float16* __restrict__ A, const _Float16* __restrict__ Bw,
                             float* __restrict__ C, int M, int N, int K) {
    int wv = (blockIdx.x * blockDim.x + threadIdx.x) >> 5;
    int tiles_n = N >> 6;
    int total = (M >> 5) * tiles_n;
    if (wv >= total) return;           // whole-wave uniform exit
    int tm = wv / tiles_n;             // 32-row tile index
    int tg = wv % tiles_n;             // 64-col group index
    int lane = threadIdx.x & 31;
    int mloc = lane & 15;
    int hi   = lane >> 4;

    // A fragment: lane -> M = mloc (+16 for 2nd tile); K=k0+16*hi+i.
    // B fragment: lane -> N=mloc, K=k0+16*hi+i.
    const _Float16* arow0 = A + (size_t)(tm * 32 + mloc) * K + 16 * hi;
    const _Float16* arow1 = arow0 + (size_t)16 * K;
    const _Float16* bbase = Bw + (size_t)(tg * 64 + mloc) * K + 16 * hi;
    const size_t bstride = (size_t)16 * K;

    v8f acc[2][4] = {};
    for (int k0 = 0; k0 < K; k0 += 32) {
        // Warm lines for k0+64 (speculative prefetch; tail overshoot is harmless).
        __builtin_prefetch(arow0 + k0 + 64, 0, 0);
        __builtin_prefetch(arow1 + k0 + 64, 0, 0);
#pragma unroll
        for (int j = 0; j < 4; j++)
            __builtin_prefetch(bbase + (size_t)j * bstride + k0 + 64, 0, 0);

        v16h a0, a1;
        {
            v8h lo0 = *(const v8h*)(arow0 + k0);
            v8h hi0 = *(const v8h*)(arow0 + k0 + 8);
            v8h lo1 = *(const v8h*)(arow1 + k0);
            v8h hi1 = *(const v8h*)(arow1 + k0 + 8);
#pragma unroll
            for (int i = 0; i < 8; i++) {
                a0[i] = lo0[i]; a0[i + 8] = hi0[i];
                a1[i] = lo1[i]; a1[i + 8] = hi1[i];
            }
        }
#pragma unroll
        for (int j = 0; j < 4; j++) {
            v16h b = *(const v16h*)(bbase + (size_t)j * bstride + k0);
            acc[0][j] = __builtin_amdgcn_wmma_f32_16x16x32_f16(
                false, a0, false, b, (short)0, acc[0][j], false, false);
            acc[1][j] = __builtin_amdgcn_wmma_f32_16x16x32_f16(
                false, a1, false, b, (short)0, acc[1][j], false, false);
        }
    }

#pragma unroll
    for (int mi = 0; mi < 2; mi++) {
#pragma unroll
        for (int j = 0; j < 4; j++) {
            int cn = tg * 64 + j * 16 + mloc;
            float* cp = C + (size_t)(tm * 32 + mi * 16 + 8 * hi) * N + cn;
#pragma unroll
            for (int r = 0; r < 8; r++)
                cp[(size_t)r * N] = acc[mi][j][r];
        }
    }
}

// ------------------------------------------------- causal depthwise conv + silu
__global__ void conv_silu(const float* __restrict__ xz, const float* __restrict__ cw,
                          const float* __restrict__ cb, float* __restrict__ xs32,
                          _Float16* __restrict__ xs16) {
    int idx = blockIdx.x * blockDim.x + threadIdx.x;
    if (idx >= ROWS * DIN) return;
    int d   = idx & (DIN - 1);
    int row = idx >> 11;
    int t   = row & (LL - 1);
    float acc = cb[d];
#pragma unroll
    for (int k = 0; k < KK; k++) {
        int tt = t - (KK - 1) + k;
        if (tt >= 0)
            acc += cw[d * KK + k] * xz[(size_t)(row - (KK - 1) + k) * (2 * DIN) + d];
    }
    float s = acc * (1.f / (1.f + __expf(-acc)));
    xs32[idx] = s;
    xs16[idx] = (_Float16)s;
}

// ------------------------------------------------- x_dbl[:, :R] -> packed f16
__global__ void extract_dt_in(const float* __restrict__ xdbl, _Float16* __restrict__ dtA) {
    int idx = blockIdx.x * blockDim.x + threadIdx.x;
    if (idx >= ROWS * RR) return;
    int row = idx >> 6, c = idx & 63;
    dtA[idx] = (_Float16)xdbl[(size_t)row * XDBL_LD + c];
}

// ------------------------------------------------- softplus(dt_raw + bias)
__global__ void dt_softplus(const float* __restrict__ raw, const float* __restrict__ bias,
                            float* __restrict__ out) {
    int idx = blockIdx.x * blockDim.x + threadIdx.x;
    if (idx >= ROWS * DIN) return;
    int d = idx & (DIN - 1);
    float v = raw[idx] + bias[d];
    out[idx] = (v > 20.f) ? v : log1pf(__expf(v));
}

// ------------------------------------------------- selective scan (sequential in t)
__global__ void scan_kernel(const float* __restrict__ dt, const float* __restrict__ xdbl,
                            const float* __restrict__ xs, const float* __restrict__ xz,
                            const float* __restrict__ A_log, const float* __restrict__ Dp,
                            _Float16* __restrict__ y16) {
    __shared__ float sB[NN], sC[NN];
    int b = blockIdx.x >> 3;
    int d = ((blockIdx.x & 7) << 8) + threadIdx.x;
    float Aa[NN], h[NN];
#pragma unroll
    for (int n = 0; n < NN; n++) { Aa[n] = -__expf(A_log[d * NN + n]); h[n] = 0.f; }
    float Dd = Dp[d];
    for (int t = 0; t < LL; t++) {
        int row = b * LL + t;
        if (threadIdx.x < NN)          sB[threadIdx.x]      = xdbl[(size_t)row * XDBL_LD + RR + threadIdx.x];
        else if (threadIdx.x < 2 * NN) sC[threadIdx.x - NN] = xdbl[(size_t)row * XDBL_LD + RR + NN + (threadIdx.x - NN)];
        __syncthreads();
        float dtv = dt[(size_t)row * DIN + d];
        float xv  = xs[(size_t)row * DIN + d];
        float dx  = dtv * xv;
        float acc = 0.f;
#pragma unroll
        for (int n = 0; n < NN; n++) {
            h[n] = h[n] * __expf(dtv * Aa[n]) + dx * sB[n];
            acc += h[n] * sC[n];
        }
        float y  = acc + xv * Dd;
        float zv = xz[(size_t)row * (2 * DIN) + DIN + d];
        float sz = zv * (1.f / (1.f + __expf(-zv)));
        y16[(size_t)row * DIN + d] = (_Float16)(y * sz);
        __syncthreads();
    }
}

// ------------------------------------------------- gelu(fc_raw + b) -> f16
__global__ void bias_gelu(const float* __restrict__ raw, const float* __restrict__ bias,
                          _Float16* __restrict__ out16) {
    int idx = blockIdx.x * blockDim.x + threadIdx.x;
    if (idx >= ROWS * 4 * EE) return;
    int f = idx & (4 * EE - 1);
    float v = raw[idx] + bias[f];
    float g = 0.5f * v * (1.f + erff(v * 0.70710678118654752f));
    out16[idx] = (_Float16)g;
}

// ------------------------------------------------- out = x2 + p + proj_b
__global__ void final_add(const float* __restrict__ x2, const float* __restrict__ p,
                          const float* __restrict__ pb, float* __restrict__ out) {
    int idx = blockIdx.x * blockDim.x + threadIdx.x;
    if (idx >= ROWS * EE) return;
    int e = idx & (EE - 1);
    out[idx] = x2[idx] + p[idx] + pb[e];
}

// =======================================================================
extern "C" void kernel_launch(void* const* d_in, const int* in_sizes, int n_in,
                              void* d_out, int out_size, void* d_ws, size_t ws_size,
                              hipStream_t stream) {
    const float* x      = (const float*)d_in[0];
    const float* ln1_w  = (const float*)d_in[1];
    const float* ln1_b  = (const float*)d_in[2];
    const float* in_w   = (const float*)d_in[3];   // (4096,1024)
    const float* conv_w = (const float*)d_in[4];   // (2048,1,4)
    const float* conv_b = (const float*)d_in[5];
    const float* xprj_w = (const float*)d_in[6];   // (96,2048)
    const float* dt_w   = (const float*)d_in[7];   // (2048,64)
    const float* dt_b   = (const float*)d_in[8];
    const float* A_log  = (const float*)d_in[9];   // (2048,16)
    const float* Dp     = (const float*)d_in[10];
    const float* out_w  = (const float*)d_in[11];  // (1024,2048)
    const float* ln2_w  = (const float*)d_in[12];
    const float* ln2_b  = (const float*)d_in[13];
    const float* fc_w   = (const float*)d_in[14];  // (4096,1024)
    const float* fc_b   = (const float*)d_in[15];
    const float* proj_w = (const float*)d_in[16];  // (1024,4096)
    const float* proj_b = (const float*)d_in[17];
    float* out = (float*)d_out;

    char* ws = (char*)d_ws;
    size_t off = 0;
    auto alloc = [&](size_t bytes) -> char* {
        char* p = ws + off;
        off = (off + bytes + 255) & ~(size_t)255;
        return p;
    };
    _Float16* w1h   = (_Float16*)alloc((size_t)4096 * 1024 * 2);
    _Float16* w2h   = (_Float16*)alloc((size_t)XDBL_LD * 2048 * 2);
    _Float16* w3h   = (_Float16*)alloc((size_t)2048 * 64 * 2);
    _Float16* w4h   = (_Float16*)alloc((size_t)1024 * 2048 * 2);
    _Float16* wfch  = (_Float16*)alloc((size_t)4096 * 1024 * 2);
    _Float16* wph   = (_Float16*)alloc((size_t)1024 * 4096 * 2);
    _Float16* h1h   = (_Float16*)alloc((size_t)ROWS * EE * 2);
    float*    xzbuf = (float*)   alloc((size_t)ROWS * 4096 * 4);
    float*    xs32  = (float*)   alloc((size_t)ROWS * DIN * 4);
    _Float16* xs16  = (_Float16*)alloc((size_t)ROWS * DIN * 2);
    float*    xdbl  = (float*)   alloc((size_t)ROWS * XDBL_LD * 4);
    _Float16* dtA   = (_Float16*)alloc((size_t)ROWS * RR * 2);
    float*    dtraw = (float*)   alloc((size_t)ROWS * DIN * 4);
    float*    dt32  = (float*)   alloc((size_t)ROWS * DIN * 4);
    _Float16* y16   = (_Float16*)alloc((size_t)ROWS * DIN * 2);
    float*    mo    = (float*)   alloc((size_t)ROWS * EE * 4);
    float*    x2    = (float*)   alloc((size_t)ROWS * EE * 4);
    _Float16* h2h   = (_Float16*)alloc((size_t)ROWS * EE * 2);
    float*    fcraw = (float*)   alloc((size_t)ROWS * 4 * EE * 4);
    _Float16* hg16  = (_Float16*)alloc((size_t)ROWS * 4 * EE * 2);
    float*    p32   = (float*)   alloc((size_t)ROWS * EE * 4);

    auto cvt = [&](const float* s, _Float16* d, int n) {
        cvt_f32_f16<<<(n + 255) / 256, 256, 0, stream>>>(s, d, n);
    };
    auto gemm = [&](const _Float16* A, const _Float16* Bw, float* C, int M, int N, int K) {
        int waves = (M / 32) * (N / 64);
        int blocks = (waves * 32 + 255) / 256;
        wmma_gemm_nt<<<blocks, 256, 0, stream>>>(A, Bw, C, M, N, K);
    };

    // weight f32 -> f16 (deterministic, recomputed each call)
    cvt(in_w,   w1h,  4096 * 1024);
    cvt_pad_w2<<<(XDBL_LD * DIN + 255) / 256, 256, 0, stream>>>(xprj_w, w2h);
    cvt(dt_w,   w3h,  2048 * 64);
    cvt(out_w,  w4h,  1024 * 2048);
    cvt(fc_w,   wfch, 4096 * 1024);
    cvt(proj_w, wph,  1024 * 4096);

    // 1) LN1
    ln_kernel<<<ROWS, 256, 0, stream>>>(x, nullptr, ln1_w, ln1_b, h1h, nullptr);
    // 2) in_proj: xz = h1 * W1^T  (2048 x 4096)
    gemm(h1h, w1h, xzbuf, ROWS, 4096, EE);
    // 3) depthwise causal conv + silu -> xs
    conv_silu<<<(ROWS * DIN + 255) / 256, 256, 0, stream>>>(xzbuf, conv_w, conv_b, xs32, xs16);
    // 4) x_proj (padded): x_dbl = xs * W2p^T (2048 x 128)
    gemm(xs16, w2h, xdbl, ROWS, XDBL_LD, DIN);
    // 5) pack dt input (first 64 cols) -> f16
    extract_dt_in<<<(ROWS * RR + 255) / 256, 256, 0, stream>>>(xdbl, dtA);
    // 6) dt_proj: dt_raw = dtA * W3^T (2048 x 2048, K=64)
    gemm(dtA, w3h, dtraw, ROWS, DIN, RR);
    // 7) softplus(+bias)
    dt_softplus<<<(ROWS * DIN + 255) / 256, 256, 0, stream>>>(dtraw, dt_b, dt32);
    // 8) selective scan (+ xs*D, * silu(z), -> f16)
    scan_kernel<<<BB * (DIN / 256), 256, 0, stream>>>(dt32, xdbl, xs32, xzbuf, A_log, Dp, y16);
    // 9) out_proj: mo = y * W4^T (2048 x 1024)
    gemm(y16, w4h, mo, ROWS, EE, DIN);
    // 10) residual + LN2 (stores x2 = x + mo)
    ln_kernel<<<ROWS, 256, 0, stream>>>(x, mo, ln2_w, ln2_b, h2h, x2);
    // 11) fc: fcraw = h2 * Wfc^T (2048 x 4096)
    gemm(h2h, wfch, fcraw, ROWS, 4 * EE, EE);
    // 12) bias + exact gelu -> f16
    bias_gelu<<<(ROWS * 4 * EE + 255) / 256, 256, 0, stream>>>(fcraw, fc_b, hg16);
    // 13) proj: p = hg * Wp^T (2048 x 1024, K=4096)
    gemm(hg16, wph, p32, ROWS, EE, 4 * EE);
    // 14) out = x2 + p + proj_b
    final_add<<<(ROWS * EE + 255) / 256, 256, 0, stream>>>(x2, p32, proj_b, out);
}
